// SparseWindowPartitionLayer_31834297598782
// MI455X (gfx1250) — compile-verified, matrix operand order checked by reading, not run on hardware
//
#include <hip/hip_runtime.h>
#include <hip/hip_bf16.h>

typedef __attribute__((ext_vector_type(16))) _Float16 v16h;
typedef __attribute__((ext_vector_type(8)))  float    v8f;

#define NVOX   300000
#define FEAT   192
#define MXW    41
#define MYW    41
#define MZW    5
#define MPER   (MXW*MYW*MZW)   /* 8405 */
#define NW     (2*MPER)        /* 16810: batch in {0,1} */

__device__ __forceinline__ int lvl_of(int c, int* tgt){
  if (c < 16){ *tgt = 16; return 0; }
  if (c < 32){ *tgt = 32; return 1; }
  *tgt = 48; return 2;
}

__device__ __forceinline__ void win_calc(int b,int z,int y,int x,
                                         int shx,int shy,int shz,
                                         int* bwi,int* ciw){
  int scx = x + shx, scy = y + shy, scz = z + shz;
  int wcx = scx/10, wcy = scy/10, wcz = scz/4;
  *bwi = b*MPER + wcx*(MYW*MZW) + wcy*MZW + wcz;
  *ciw = (scz%4) | ((scy%10)<<8) | ((scx%10)<<16);
}

// ---------------- pass 0: init scratch ----------------
__global__ void k_init(int* hist0,int* hist1,int* cnt2,int* winslot,
                       int* nkept,int* nwinlvl){
  int i = blockIdx.x*blockDim.x + threadIdx.x;
  if (i < NW){ hist0[i] = 0; hist1[i] = 0; }
  if (i < 2*NW){ cnt2[i] = 0; winslot[i] = -1; }
  if (i < 6) nwinlvl[i] = 0;
  if (i == 0) nkept[0] = 0;
}

// ---------------- pass 1: window inds + hist0 ----------------
__global__ void k_pass1(const int* __restrict__ coords, int* bwi, int* ciwp,
                        int* rank0, int* hist0){
  int v = blockIdx.x*blockDim.x + threadIdx.x;
  if (v >= NVOX) return;
  if (v + 2048 < NVOX) __builtin_prefetch(coords + (v+2048)*4, 0, 0);
  int b = coords[v*4+0], z = coords[v*4+1], y = coords[v*4+2], x = coords[v*4+3];
  int w0,c0,w1,c1;
  win_calc(b,z,y,x,10,10,4,&w0,&c0);   // shift=False: sh = window_shape
  win_calc(b,z,y,x, 5, 5,2,&w1,&c1);   // shift=True : sh = window_shape/2
  bwi[v]        = w0;  bwi[NVOX+v]  = w1;
  ciwp[v]       = c0;  ciwp[NVOX+v] = c1;
  rank0[v] = atomicAdd(&hist0[w0], 1);
}

// ---------------- pass 2: keep0 + hist1 ----------------
__global__ void k_pass2(const int* __restrict__ bwi, const int* __restrict__ rank0,
                        int* rank1, const int* __restrict__ hist0, int* hist1){
  int v = blockIdx.x*blockDim.x + threadIdx.x;
  if (v >= NVOX) return;
  int t0; (void)lvl_of(hist0[bwi[v]], &t0);
  if (rank0[v] < t0) rank1[v] = atomicAdd(&hist1[bwi[NVOX+v]], 1);
  else               rank1[v] = -1;
}

// ---------------- pass 3: final keep, per-window final rank, slot assign ----
__global__ void k_pass3(const int* __restrict__ bwi, const int* __restrict__ rank0,
                        const int* __restrict__ rank1,
                        const int* __restrict__ hist0, const int* __restrict__ hist1,
                        int* cnt2, int* winslot, int* nkept, int* nwinlvl,
                        int* keptIdx, int* keptRank, int* keptLvl){
  int v = blockIdx.x*blockDim.x + threadIdx.x;
  if (v >= NVOX) return;
  int w0 = bwi[v], w1 = bwi[NVOX+v];
  int t0; int l0 = lvl_of(hist0[w0], &t0);
  if (rank0[v] >= t0) return;
  int r1 = rank1[v];
  int t1; int l1 = lvl_of(hist1[w1], &t1);
  if (r1 < 0 || r1 >= t1) return;
  int pos = atomicAdd(nkept, 1);
  keptIdx[pos] = v;
  int wss[2] = {w0, w1};
  int lss[2] = {l0, l1};
  #pragma unroll
  for (int s = 0; s < 2; ++s){
    int fr = atomicAdd(&cnt2[s*NW + wss[s]], 1);
    keptRank[s*NVOX + pos] = fr;
    keptLvl [s*NVOX + pos] = lss[s];
    if (fr == 0){
      int slot = atomicAdd(&nwinlvl[s*3 + lss[s]], 1);
      winslot[s*NW + wss[s]] = slot;
    }
  }
}

// ---------------- pass 4: dest rows ----------------
__global__ void k_pass4(const int* __restrict__ bwi, const int* __restrict__ winslot,
                        const int* __restrict__ nkept, const int* __restrict__ keptIdx,
                        const int* __restrict__ keptRank, const int* __restrict__ keptLvl,
                        int* keptRow){
  int pos = blockIdx.x*blockDim.x + threadIdx.x;
  if (pos >= nkept[0]) return;
  int v = keptIdx[pos];
  #pragma unroll
  for (int s = 0; s < 2; ++s){
    int w    = bwi[s*NVOX + v];
    int slot = winslot[s*NW + w];
    int lvl  = keptLvl[s*NVOX + pos];
    int T    = (lvl == 0) ? 16 : (lvl == 1) ? 32 : 48;
    keptRow[s*NVOX + pos] = slot*T + keptRank[s*NVOX + pos];
  }
}

// ---------------- pass 5: output offset table ----------------
// layout: feat | coord | pe[0][0..2] | mask[0][0..2] | pe[1][0..2] | mask[1][0..2]
// offs: [0]=feat [1]=coord [2+s*6+d]=pe [5+s*6+d]=mask [14]=end
__global__ void k_offsets(const int* __restrict__ nkept, const int* __restrict__ nwinlvl,
                          long long* offs){
  if (threadIdx.x != 0 || blockIdx.x != 0) return;
  long long K = nkept[0];
  int T[3] = {16, 32, 48};
  long long o = 0;
  offs[0] = 0;        o  = K * FEAT;
  offs[1] = o;        o += K * 4;
  for (int s = 0; s < 2; ++s){
    for (int d = 0; d < 3; ++d){ offs[2 + s*6 + d] = o; o += (long long)nwinlvl[s*3+d] * T[d] * FEAT; }
    for (int d = 0; d < 3; ++d){ offs[5 + s*6 + d] = o; o += (long long)nwinlvl[s*3+d] * T[d]; }
  }
  offs[14] = o;
}

// ---------------- pass 6: zero output, masks default to 1 (not filled) ------
__global__ void k_zero(float* __restrict__ out, long long total,
                       const long long* __restrict__ offs){
  long long i = (long long)blockIdx.x*blockDim.x + threadIdx.x;
  if (i >= total) return;
  float v = 0.0f;
  if ((i >= offs[5] && i < offs[8]) || (i >= offs[11] && i < offs[14])) v = 1.0f;
  out[i] = v;
}

// ---------------- pass 7: gather kept features / coords ----------------
__global__ void k_feats(const float* __restrict__ feats, const int* __restrict__ keptIdx,
                        const int* __restrict__ nkept, float* __restrict__ out){
  long long i = (long long)blockIdx.x*blockDim.x + threadIdx.x;
  if (i >= (long long)NVOX*FEAT) return;
  int pos = (int)(i / FEAT), c = (int)(i % FEAT);
  if (pos >= nkept[0]) return;
  __builtin_prefetch(feats + (long long)keptIdx[pos]*FEAT + c, 0, 0);
  out[i] = feats[(long long)keptIdx[pos]*FEAT + c];
}

__global__ void k_coords(const int* __restrict__ coords, const int* __restrict__ keptIdx,
                         const int* __restrict__ nkept, const long long* __restrict__ offs,
                         float* __restrict__ out){
  int i = blockIdx.x*blockDim.x + threadIdx.x;
  if (i >= NVOX*4) return;
  int pos = i >> 2;
  if (pos >= nkept[0]) return;
  out[offs[1] + i] = (float)coords[keptIdx[pos]*4 + (i & 3)];
}

// ---------------- pass 8: WMMA positional embedding + scatter ----------------
// One wave handles (shift s, group of 16 kept voxels).
// Phase matrix e[m][n] = v[m] * inv_freq[n] via rank-1 WMMA:
//   A 16x32 f16 : A[m][0] = v[m], else 0  (lane m<16, half element 0)
//   B 32x16 f16 : B[0][n] = inv_freq[n]  (lane n<16, half element 0)
//   D = A*B -> 16x16 f32, lane L reg r -> (M = r + 8*(L>=16), N = L&15)
// Stores are branchless (invalid rows -> dump slot); cos(x)=sin(x+pi/2);
// tile loop kept rolled (#pragma unroll 1) so only one 3-WMMA group of
// accumulators is live -> no VGPR spills.
__global__ void k_posembed(const int* __restrict__ ciwp, const int* __restrict__ keptIdx,
                           const int* __restrict__ keptLvl, const int* __restrict__ keptRow,
                           const int* __restrict__ nkept, const long long* __restrict__ offs,
                           float* __restrict__ dump, float* __restrict__ out){
  int tid  = blockIdx.x*blockDim.x + threadIdx.x;
  int wave = tid >> 5;
  int lane = tid & 31;
  int s    = wave & 1;
  int g    = wave >> 1;
  int K    = nkept[0];
  int base = g * 16;
  if (base >= K) return;                 // wave-uniform exit; EXEC stays full
  int  lane16 = lane & 15;
  bool lo     = lane < 16;
  int  pos    = base + lane16;
  bool valid  = lo && (pos < K);

  float vx = 0.f, vy = 0.f, vz = 0.f;
  long long myBase = -1, myMask = -1;
  if (valid){
    int v   = keptIdx[pos];
    int p   = ciwp[s*NVOX + v];
    vz = (float)( p        & 255) - 2.0f;   // z - wz/2
    vy = (float)((p >> 8)  & 255) - 5.0f;   // y - wy/2
    vx = (float)((p >> 16) & 255) - 5.0f;   // x - wx/2
    int lvl = keptLvl[s*NVOX + pos];
    int row = keptRow[s*NVOX + pos];
    myBase = offs[2 + s*6 + lvl] + (long long)row * FEAT;
    myMask = offs[5 + s*6 + lvl] + row;
  }
  // mask: filled -> key_padding_mask false (branchless via dump redirect)
  { float* mp = valid ? (out + myMask) : (dump + lane);
    *mp = 0.0f; }

  // Broadcast per-row output pointers to all lanes (ds_bpermute), pre-resolved
  // against the dump slot so the inner stores need no branches at all.
  float* rowPtr[8];
  #pragma unroll
  for (int r = 0; r < 8; ++r){
    int m = r + ((lane >> 4) << 3);
    long long bb = __shfl(myBase, m, 32);
    rowPtr[r] = (bb >= 0) ? (out + bb) : (dump + lane);
  }

  v16h Ax = {}, Ay = {}, Az = {};
  Ax[0] = (_Float16)vx;                   // lanes>=16 / invalid carry 0
  Ay[0] = (_Float16)vy;
  Az[0] = (_Float16)vz;

  const float L2K = 9.965784284662087f;   // log2(1000)
  #pragma unroll 1
  for (int t = 0; t < 4; ++t){
    int   jj   = t*16 + lane16;
    float fr   = exp2f(-L2K * (float)(2*(jj >> 1)) * (1.0f/64.0f)); // 1/inv_freq
    float poff = (jj & 1) ? 1.5707963267948966f : 0.0f;  // cos(x)=sin(x+pi/2)
    v16h  B    = {};
    B[0] = lo ? (_Float16)fr : (_Float16)0.0f;
    v8f cx = {}, cy = {}, cz = {};
    cx = __builtin_amdgcn_wmma_f32_16x16x32_f16(false, Ax, false, B, (short)0, cx, false, false);
    cy = __builtin_amdgcn_wmma_f32_16x16x32_f16(false, Ay, false, B, (short)0, cy, false, false);
    cz = __builtin_amdgcn_wmma_f32_16x16x32_f16(false, Az, false, B, (short)0, cz, false, false);
    #pragma unroll
    for (int r = 0; r < 8; ++r){
      float* p = rowPtr[r] + jj;          // one 64-bit addr calc per row
      p[0]   = __sinf(cx[r] + poff);      // x segment: cols   0..63
      p[64]  = __sinf(cy[r] + poff);      // y segment: cols  64..127
      p[128] = __sinf(cz[r] + poff);      // z segment: cols 128..191
    }
  }
}

extern "C" void kernel_launch(void* const* d_in, const int* in_sizes, int n_in,
                              void* d_out, int out_size, void* d_ws, size_t ws_size,
                              hipStream_t stream) {
  const float* feats  = (const float*)d_in[0];
  const int*   coords = (const int*)d_in[1];
  float* out = (float*)d_out;

  char* w = (char*)d_ws;
  long long* offs = (long long*)w;            w += 16*sizeof(long long);
  float* dump   = (float*)w;                  w += 256*sizeof(float);
  int* bwi      = (int*)w;                    w += 2*NVOX*sizeof(int);
  int* ciwp     = (int*)w;                    w += 2*NVOX*sizeof(int);
  int* rank0    = (int*)w;                    w += NVOX*sizeof(int);
  int* rank1    = (int*)w;                    w += NVOX*sizeof(int);
  int* hist0    = (int*)w;                    w += NW*sizeof(int);
  int* hist1    = (int*)w;                    w += NW*sizeof(int);
  int* cnt2     = (int*)w;                    w += 2*NW*sizeof(int);
  int* winslot  = (int*)w;                    w += 2*NW*sizeof(int);
  int* nkept    = (int*)w;                    w += sizeof(int);
  int* nwinlvl  = (int*)w;                    w += 8*sizeof(int);
  int* keptIdx  = (int*)w;                    w += NVOX*sizeof(int);
  int* keptRank = (int*)w;                    w += 2*NVOX*sizeof(int);
  int* keptLvl  = (int*)w;                    w += 2*NVOX*sizeof(int);
  int* keptRow  = (int*)w;                    w += 2*NVOX*sizeof(int);

  const int thr = 256;
  k_init   <<<(2*NW + thr-1)/thr, thr, 0, stream>>>(hist0,hist1,cnt2,winslot,nkept,nwinlvl);
  k_pass1  <<<(NVOX + thr-1)/thr, thr, 0, stream>>>(coords,bwi,ciwp,rank0,hist0);
  k_pass2  <<<(NVOX + thr-1)/thr, thr, 0, stream>>>(bwi,rank0,rank1,hist0,hist1);
  k_pass3  <<<(NVOX + thr-1)/thr, thr, 0, stream>>>(bwi,rank0,rank1,hist0,hist1,
                                                    cnt2,winslot,nkept,nwinlvl,
                                                    keptIdx,keptRank,keptLvl);
  k_pass4  <<<(NVOX + thr-1)/thr, thr, 0, stream>>>(bwi,winslot,nkept,keptIdx,
                                                    keptRank,keptLvl,keptRow);
  k_offsets<<<1, 1, 0, stream>>>(nkept,nwinlvl,offs);
  k_zero   <<<(out_size + thr-1)/thr, thr, 0, stream>>>(out,(long long)out_size,offs);
  long long featWork = (long long)NVOX*FEAT;
  k_feats  <<<(int)((featWork + thr-1)/thr), thr, 0, stream>>>(feats,keptIdx,nkept,out);
  k_coords <<<(NVOX*4 + thr-1)/thr, thr, 0, stream>>>(coords,keptIdx,nkept,offs,out);
  int waves  = 2 * ((NVOX + 15)/16);
  int blocks = (waves*32 + thr-1)/thr;
  k_posembed<<<blocks, thr, 0, stream>>>(ciwp,keptIdx,keptLvl,keptRow,nkept,offs,dump,out);
}
